// CompoundPoissonQKV_69836168233137
// MI455X (gfx1250) — matrix operand order repaired and checked by reading, not compiled, not run on hardware
//
#include <hip/hip_runtime.h>

typedef __attribute__((ext_vector_type(16))) __bf16 v16bf;
typedef __attribute__((ext_vector_type(8)))  float   v8f;

constexpr int   B_ = 4, N_ = 4096, D_ = 256, TOPK_ = 49;
constexpr float SCALE_  = 0.0625f;   // 1/sqrt(256)
constexpr float LN_EPS_ = 1e-5f;

// ---------------- workspace layout (bytes) ----------------
constexpr size_t O_XB   = 0;                                      // X bf16            [B,N,D]
constexpr size_t O_WQT  = O_XB   + (size_t)B_*N_*D_*2;            // W_Q^T bf16        [D,D]
constexpr size_t O_WKT  = O_WQT  + (size_t)D_*D_*2;               // W_K^T bf16        [D,D]
constexpr size_t O_WVGT = O_WKT  + (size_t)D_*D_*2;               // (W_V@gcn)^T bf16  [B,D,D]
constexpr size_t O_QN   = O_WVGT + (size_t)B_*D_*D_*2;            // l2norm Q bf16     [B,N,D]
constexpr size_t O_KN   = O_QN   + (size_t)B_*N_*D_*2;            // l2norm K bf16     [B,N,D]
constexpr size_t O_SUP  = O_KN   + (size_t)B_*N_*D_*2;            // support f32       [B,N,D]
constexpr size_t O_GCN  = O_SUP  + (size_t)B_*N_*D_*4;            // gcn_out bf16      [B,N,D]
constexpr size_t O_GT   = O_GCN  + (size_t)B_*N_*D_*2;            // gcn_out^T bf16    [B,D,N]
constexpr size_t O_DOTS = O_GT   + (size_t)B_*N_*D_*2;            // dots f32          [B,N,N]
constexpr size_t O_PROB = O_DOTS + (size_t)B_*N_*(size_t)N_*4;    // softmax bf16      [B,N,N]

// ---------------- helpers ----------------
__device__ inline unsigned short f32_to_bf16(float f) {
  union { float f; unsigned u; } v; v.f = f;
  unsigned u = v.u;
  u += 0x7fffu + ((u >> 16) & 1u);           // round-to-nearest-even
  return (unsigned short)(u >> 16);
}

union Frag16 { uint4 q[2]; v16bf v; };

// A fragment, 16x32 bf16, row-major [*, lda]; contiguous k order (same as load_frag_BT):
// row m = lane&15 ; element j -> K = k0 + (lane>=16?16:0) + j
__device__ inline v16bf load_frag_A(const __bf16* p, int lda, int row0, int k0, int lane) {
  const __bf16* rp = p + (size_t)(row0 + (lane & 15)) * lda + k0 + ((lane >> 4) << 4);
  Frag16 f;
  f.q[0] = *reinterpret_cast<const uint4*>(rp);        // K j=0..7 (8 bf16, 16B)
  f.q[1] = *reinterpret_cast<const uint4*>(rp + 8);    // K j=8..15
  return f.v;
}

// B fragment, 32x16 bf16, where column n of B is stored as a contiguous row
// (transposed operand): lane holds col n = col0+lane&15 ; K = k0 + (lane>=16?16:0) + j
__device__ inline v16bf load_frag_BT(const __bf16* p, int lda, int col0, int k0, int lane) {
  const __bf16* rp = p + (size_t)(col0 + (lane & 15)) * lda + k0 + ((lane >> 4) << 4);
  Frag16 f;
  f.q[0] = *reinterpret_cast<const uint4*>(rp);        // 16 contiguous bf16 (32B)
  f.q[1] = *reinterpret_cast<const uint4*>(rp + 8);
  return f.v;
}

__device__ inline float blk_sum(float v, float* red, int tid) {
  red[tid] = v; __syncthreads();
#pragma unroll
  for (int s = 128; s > 0; s >>= 1) {
    if (tid < s) red[tid] += red[tid + s];
    __syncthreads();
  }
  float r = red[0]; __syncthreads();
  return r;
}
__device__ inline float blk_max(float v, float* red, int tid) {
  red[tid] = v; __syncthreads();
#pragma unroll
  for (int s = 128; s > 0; s >>= 1) {
    if (tid < s) red[tid] = fmaxf(red[tid], red[tid + s]);
    __syncthreads();
  }
  float r = red[0]; __syncthreads();
  return r;
}

// ---------------- kernels ----------------
__global__ void __launch_bounds__(256) f2bf_kernel(const float* src, unsigned short* dst, int count) {
  int i = blockIdx.x * 256 + threadIdx.x;
  if (i < count) dst[i] = f32_to_bf16(src[i]);
}

// 256x256 f32 -> bf16 transposed: dst[n*D+k] = src[k*D+n]
__global__ void __launch_bounds__(256) wtrans_kernel(const float* src, unsigned short* dst) {
  const int n = blockIdx.x, k = threadIdx.x;
  dst[n * D_ + k] = f32_to_bf16(src[k * D_ + n]);
}

// WVGt[b][d][k] = sum_j W_V[k][j] * gcn_w[b][j][d]   (transposed product, coalesced stores)
__global__ void __launch_bounds__(256) wvg_kernel(const float* WV, const float* GW, unsigned short* WVGt) {
  const int b = blockIdx.x >> 8, d = blockIdx.x & 255, k = threadIdx.x;
  const float* g = GW + (size_t)b * D_ * D_;
  float acc = 0.f;
  for (int j = 0; j < D_; ++j)
    acc = fmaf(WV[k * D_ + j], g[j * D_ + d], acc);
  WVGt[((size_t)b * D_ + d) * D_ + k] = f32_to_bf16(acc);
}

// Q = l2norm(X Wq), K = l2norm(X Wk), support = X (Wv G_b); one 16-row tile per block.
__global__ void __launch_bounds__(256) qkv_kernel(const unsigned short* Xb_,
                                                  const unsigned short* WQt_, const unsigned short* WKt_,
                                                  const unsigned short* WVGt_,
                                                  unsigned short* Qn_, unsigned short* Kn_, float* Sup_) {
  __shared__ float sQ[16 * 256];
  __shared__ float sK[16 * 256];
  __shared__ float sInv[32];
  const int tid = threadIdx.x, lane = tid & 31, w = tid >> 5;
  const int b = blockIdx.x >> 8;
  const int row0 = (blockIdx.x & 255) << 4;
  const __bf16* Xp = (const __bf16*)Xb_ + (size_t)b * N_ * D_;
  const __bf16* WQ = (const __bf16*)WQt_;
  const __bf16* WK = (const __bf16*)WKt_;
  const __bf16* WG = (const __bf16*)WVGt_ + (size_t)b * D_ * D_;
  v8f accQ[2] = {}, accK[2] = {}, accS[2] = {};
  for (int kk = 0; kk < D_; kk += 32) {
    v16bf a = load_frag_A(Xp, D_, row0, kk, lane);
#pragma unroll
    for (int j = 0; j < 2; ++j) {
      const int col0 = ((w << 1) + j) << 4;
      v16bf bq = load_frag_BT(WQ, D_, col0, kk, lane);
      v16bf bk = load_frag_BT(WK, D_, col0, kk, lane);
      v16bf bs = load_frag_BT(WG, D_, col0, kk, lane);
      accQ[j] = __builtin_amdgcn_wmma_f32_16x16x32_bf16(false, a, false, bq, (short)0, accQ[j], false, false);
      accK[j] = __builtin_amdgcn_wmma_f32_16x16x32_bf16(false, a, false, bk, (short)0, accK[j], false, false);
      accS[j] = __builtin_amdgcn_wmma_f32_16x16x32_bf16(false, a, false, bs, (short)0, accS[j], false, false);
    }
  }
#pragma unroll
  for (int j = 0; j < 2; ++j) {
    const int coln = ((((w << 1) + j) << 4)) + (lane & 15);
#pragma unroll
    for (int r = 0; r < 8; ++r) {
      const int m = r + ((lane >> 4) << 3);
      sQ[m * 256 + coln] = accQ[j][r];
      sK[m * 256 + coln] = accK[j][r];
      Sup_[((size_t)b * N_ + row0 + m) * D_ + coln] = accS[j][r];
    }
  }
  __syncthreads();
  if (tid < 32) {
    const float* rp = (tid < 16) ? (sQ + tid * 256) : (sK + (tid - 16) * 256);
    float ss = 0.f;
    for (int i = 0; i < 256; ++i) { float v = rp[i]; ss = fmaf(v, v, ss); }
    sInv[tid] = 1.0f / fmaxf(sqrtf(ss), 1e-12f);
  }
  __syncthreads();
#pragma unroll
  for (int i = 0; i < 16; ++i) {
    const size_t o = ((size_t)b * N_ + row0 + i) * D_ + tid;
    Qn_[o] = f32_to_bf16(sQ[i * 256 + tid] * sInv[i]);
    Kn_[o] = f32_to_bf16(sK[i * 256 + tid] * sInv[16 + i]);
  }
}

// dots = (Qn Kn^T) * scale ; one 16x256 tile per block
__global__ void __launch_bounds__(256) dots_kernel(const unsigned short* Qn_, const unsigned short* Kn_,
                                                   float* Dots_) {
  const int tid = threadIdx.x, lane = tid & 31, w = tid >> 5;
  const int b = blockIdx.x >> 12;
  const int rem = blockIdx.x & 4095;
  const int row0 = (rem >> 4) << 4;
  const int col0 = (rem & 15) << 8;
  const __bf16* Q = (const __bf16*)Qn_ + (size_t)b * N_ * D_;
  const __bf16* K = (const __bf16*)Kn_ + (size_t)b * N_ * D_;
  v8f acc[2] = {};
  for (int kk = 0; kk < D_; kk += 32) {
    v16bf a = load_frag_A(Q, D_, row0, kk, lane);
#pragma unroll
    for (int j = 0; j < 2; ++j) {
      v16bf bb = load_frag_BT(K, D_, col0 + (((w << 1) + j) << 4), kk, lane);
      acc[j] = __builtin_amdgcn_wmma_f32_16x16x32_bf16(false, a, false, bb, (short)0, acc[j], false, false);
    }
  }
  float* drow = Dots_ + (size_t)b * N_ * (size_t)N_;
#pragma unroll
  for (int j = 0; j < 2; ++j) {
    const int coln = col0 + ((((w << 1) + j) << 4)) + (lane & 15);
#pragma unroll
    for (int r = 0; r < 8; ++r) {
      const int m = r + ((lane >> 4) << 3);
      drow[(size_t)(row0 + m) * N_ + coln] = acc[j][r] * SCALE_;
    }
  }
}

// per-row: top-49 threshold -> sparse adj @ support -> LN+ReLU -> gcn(bf16); softmax(dots*scale)->probs(bf16)
__global__ void __launch_bounds__(256) topk_softmax_gcn_kernel(const float* Dots_, const float* Sup_,
                                                               const float* lng, const float* lnb,
                                                               unsigned short* Prob_, unsigned short* Gcn_) {
  __shared__ float red[256];
  __shared__ float lval[64];
  __shared__ int   lidx[64];
  __shared__ int   cnt;
  const int tid = threadIdx.x;
  const int b = blockIdx.x >> 12;
  const int n = blockIdx.x & 4095;
  const float* row = Dots_ + ((size_t)b * N_ + n) * (size_t)N_;
  float x[16];
#pragma unroll
  for (int i = 0; i < 16; ++i) x[i] = row[tid + (i << 8)];
  float mx = -3.4e38f, mn = 3.4e38f;
#pragma unroll
  for (int i = 0; i < 16; ++i) { mx = fmaxf(mx, x[i]); mn = fminf(mn, x[i]); }
  mx = blk_max(mx, red, tid);
  mn = -blk_max(-mn, red, tid);
  // binary search for the 49th-largest value
  float lo = mn, hi = mx;
  for (int it = 0; it < 24; ++it) {
    const float mid = 0.5f * (lo + hi);
    float c = 0.f;
#pragma unroll
    for (int i = 0; i < 16; ++i) c += (x[i] >= mid) ? 1.f : 0.f;
    c = blk_sum(c, red, tid);
    if (c >= (float)TOPK_) lo = mid; else hi = mid;
  }
  const float thr = lo;
  if (tid == 0) cnt = 0;
  __syncthreads();
#pragma unroll
  for (int i = 0; i < 16; ++i) {
    if (x[i] >= thr) {
      const int p = atomicAdd(&cnt, 1);
      if (p < 64) { lval[p] = x[i]; lidx[p] = tid + (i << 8); }
    }
  }
  __syncthreads();
  int m = cnt; if (m > 64) m = 64;
  float acc = 0.f;
  const float* supb = Sup_ + (size_t)b * N_ * D_;
  for (int p = 0; p < m; ++p)
    acc = fmaf(lval[p], supb[(size_t)lidx[p] * D_ + tid], acc);   // coalesced in tid
  // softmax over dots*scale (dots already carries one scale; faithful double-scale)
  float s = 0.f; float e[16];
#pragma unroll
  for (int i = 0; i < 16; ++i) { e[i] = __expf((x[i] - mx) * SCALE_); s += e[i]; }
  s = blk_sum(s, red, tid);
  const float invs = 1.0f / s;
  unsigned short* prow = Prob_ + ((size_t)b * N_ + n) * (size_t)N_;
#pragma unroll
  for (int i = 0; i < 16; ++i) prow[tid + (i << 8)] = f32_to_bf16(e[i] * invs);
  // layernorm + relu across D (thread == channel)
  const float mu = blk_sum(acc, red, tid) * (1.f / 256.f);
  const float dd = acc - mu;
  const float var = blk_sum(dd * dd, red, tid) * (1.f / 256.f);
  float y = dd * rsqrtf(var + LN_EPS_) * lng[b * D_ + tid] + lnb[b * D_ + tid];
  y = fmaxf(y, 0.f);
  Gcn_[((size_t)b * N_ + n) * D_ + tid] = f32_to_bf16(y);
}

// LDS-tiled transpose: Gt[b][d][n] = Gcn[b][n][d]  (32x32 bf16 tiles, coalesced both sides)
__global__ void __launch_bounds__(256) gtrans_kernel(const unsigned short* Gcn_, unsigned short* Gt_) {
  __shared__ unsigned short tile[32][33];
  const int tid = threadIdx.x;
  const int b  = blockIdx.x >> 10;              // (N/32)*(D/32) = 128*8 = 1024 per batch
  const int rem = blockIdx.x & 1023;
  const int n0 = (rem >> 3) << 5;               // 128 row-tiles
  const int d0 = (rem & 7) << 5;                // 8 col-tiles
  const int c = tid & 31, r = tid >> 5;         // 32 x 8 threads
#pragma unroll
  for (int i = 0; i < 4; ++i) {
    const int n = n0 + r + (i << 3);
    tile[r + (i << 3)][c] = Gcn_[((size_t)b * N_ + n) * D_ + d0 + c];
  }
  __syncthreads();
#pragma unroll
  for (int i = 0; i < 4; ++i) {
    const int d = d0 + r + (i << 3);
    Gt_[((size_t)b * D_ + d) * N_ + n0 + c] = tile[c][r + (i << 3)];
  }
}

// attention_output = probs @ gcn ; one 16x256 tile per block, K-loop over N, B from Gt (contiguous)
__global__ void __launch_bounds__(256) attn_kernel(const unsigned short* Prob_, const unsigned short* Gt_,
                                                   float* Out_) {
  const int tid = threadIdx.x, lane = tid & 31, w = tid >> 5;
  const int b = blockIdx.x >> 8;
  const int row0 = (blockIdx.x & 255) << 4;
  const __bf16* P = (const __bf16*)Prob_ + (size_t)b * N_ * (size_t)N_;
  const __bf16* G = (const __bf16*)Gt_ + (size_t)b * D_ * N_;    // [D][N]
  v8f acc[2] = {};
  for (int kk = 0; kk < N_; kk += 32) {
    v16bf a = load_frag_A(P, N_, row0, kk, lane);
#pragma unroll
    for (int j = 0; j < 2; ++j) {
      v16bf bb = load_frag_BT(G, N_, (((w << 1) + j) << 4), kk, lane);
      acc[j] = __builtin_amdgcn_wmma_f32_16x16x32_bf16(false, a, false, bb, (short)0, acc[j], false, false);
    }
  }
#pragma unroll
  for (int j = 0; j < 2; ++j) {
    const int coln = ((((w << 1) + j) << 4)) + (lane & 15);
#pragma unroll
    for (int r = 0; r < 8; ++r) {
      const int m = r + ((lane >> 4) << 3);
      Out_[((size_t)b * N_ + row0 + m) * D_ + coln] = acc[j][r];
    }
  }
}

// ---------------- launcher ----------------
extern "C" void kernel_launch(void* const* d_in, const int* in_sizes, int n_in,
                              void* d_out, int out_size, void* d_ws, size_t ws_size,
                              hipStream_t stream) {
  (void)in_sizes; (void)n_in; (void)out_size; (void)ws_size;
  const float* X   = (const float*)d_in[0];
  const float* WQ  = (const float*)d_in[1];
  const float* WK  = (const float*)d_in[2];
  const float* WV  = (const float*)d_in[3];
  const float* GW  = (const float*)d_in[4];
  const float* LNG = (const float*)d_in[5];
  const float* LNB = (const float*)d_in[6];
  char* ws = (char*)d_ws;
  unsigned short* Xb   = (unsigned short*)(ws + O_XB);
  unsigned short* WQt  = (unsigned short*)(ws + O_WQT);
  unsigned short* WKt  = (unsigned short*)(ws + O_WKT);
  unsigned short* WVGt = (unsigned short*)(ws + O_WVGT);
  unsigned short* Qn   = (unsigned short*)(ws + O_QN);
  unsigned short* Kn   = (unsigned short*)(ws + O_KN);
  float*          Sup  = (float*)(ws + O_SUP);
  unsigned short* Gcn  = (unsigned short*)(ws + O_GCN);
  unsigned short* Gt   = (unsigned short*)(ws + O_GT);
  float*          Dots = (float*)(ws + O_DOTS);
  unsigned short* Prob = (unsigned short*)(ws + O_PROB);

  f2bf_kernel<<<(B_ * N_ * D_ + 255) / 256, 256, 0, stream>>>(X, Xb, B_ * N_ * D_);
  wtrans_kernel<<<D_, 256, 0, stream>>>(WQ, WQt);
  wtrans_kernel<<<D_, 256, 0, stream>>>(WK, WKt);
  wvg_kernel<<<B_ * D_, 256, 0, stream>>>(WV, GW, WVGt);
  qkv_kernel<<<B_ * (N_ / 16), 256, 0, stream>>>(Xb, WQt, WKt, WVGt, Qn, Kn, Sup);
  dots_kernel<<<B_ * (N_ / 16) * (N_ / 256), 256, 0, stream>>>(Qn, Kn, Dots);
  topk_softmax_gcn_kernel<<<B_ * N_, 256, 0, stream>>>(Dots, Sup, LNG, LNB, Prob, Gcn);
  gtrans_kernel<<<B_ * (N_ / 32) * (D_ / 32), 256, 0, stream>>>(Gcn, Gt);
  attn_kernel<<<B_ * (N_ / 16), 256, 0, stream>>>(Prob, Gt, (float*)d_out);
}
